// SwinBlock_9079560864361
// MI455X (gfx1250) — matrix operand order repaired and checked by reading, not compiled, not run on hardware
//
#include <hip/hip_runtime.h>
#include <hip/hip_bf16.h>

// ---------------- static config (matches reference) ----------------
#define BIMG   32
#define HIMG   64
#define WIMG   64
#define CDIM   256
#define WSZ    8
#define SHIFTV 4
#define NHEAD  8
#define HDIM   32           // head dim
#define NTOK   64           // tokens per window
#define NWIN   64           // windows per image
#define BW     (BIMG*NWIN)  // 2048 total windows
#define MROWS  (BW*NTOK)    // 131072 token rows

typedef unsigned short u16;
typedef unsigned int   u32;
typedef __attribute__((ext_vector_type(4)))  u32    u32x4;
typedef __attribute__((ext_vector_type(8)))  float  v8f;
typedef __attribute__((ext_vector_type(16))) __bf16 v16bf;

// POD fragment: 16 bf16 = 8 VGPRs, addressable as two 16B chunks
union Frag { v16bf bf; u16 u[16]; u32x4 q[2]; };
union Pack8 { u16 h[8]; u32x4 q; };

__device__ __forceinline__ u16 f2bf(float x) {   // RNE float->bf16
    u32 u = __float_as_uint(x);
    u += 0x7FFFu + ((u >> 16) & 1u);
    return (u16)(u >> 16);
}
__device__ __forceinline__ v8f vz8() {
    v8f z;
#pragma unroll
    for (int i = 0; i < 8; ++i) z[i] = 0.0f;
    return z;
}
__device__ __forceinline__ int reg3(int z) {     // shifted-window region id
    return (z < HIMG - WSZ) ? 0 : ((z < HIMG - SHIFTV) ? 1 : 2);
}

// ---------------- workspace layout (u16 elements) ----------------
#define OFF_WQKV  ((size_t)0)                    // 256x768  bf16
#define OFF_WPROJ ((size_t)196608)               // 256x256
#define OFF_WFC1  ((size_t)262144)               // 256x1024
#define OFF_WFC2  ((size_t)524288)               // 1024x256
#define OFF_XW    ((size_t)1048576)              // 131072x256 (LN1 win) / attn_out
#define OFF_Q     (OFF_XW + (size_t)MROWS*CDIM)  // q [win,h][tok][d]  / LN2 out
#define OFF_K     (OFF_Q  + (size_t)MROWS*CDIM)  // k [win,h][d][tok]
#define OFF_V     (OFF_K  + (size_t)MROWS*CDIM)  // v [win,h][tok][d]
#define OFF_G1    (OFF_V  + (size_t)MROWS*CDIM)  // 131072x1024 gelu(fc1)

// ---------------- fp32 -> bf16 weight conversion ----------------
__global__ __launch_bounds__(256) void cvt_bf16_kernel(const float* __restrict__ src,
                                                       u16* __restrict__ dst) {
    size_t i = (size_t)blockIdx.x * 256 + threadIdx.x;
    dst[i] = f2bf(src[i]);
}

// ---------------- LayerNorm (1 wave per token); DSTMODE 1 fuses shift+window ----------------
template <int DSTMODE>
__global__ __launch_bounds__(256) void ln_kernel(const float* __restrict__ in,
                                                 const float* __restrict__ g,
                                                 const float* __restrict__ bparm,
                                                 u16* __restrict__ out) {
    const int wave = threadIdx.x >> 5, lane = threadIdx.x & 31;
    const int t = blockIdx.x * 8 + wave;                 // token index
    const float* px = in + (size_t)t * CDIM + lane * 8;
    float4 a0 = *(const float4*)px;
    float4 a1 = *(const float4*)(px + 4);
    float v[8] = {a0.x, a0.y, a0.z, a0.w, a1.x, a1.y, a1.z, a1.w};
    float s = 0.f, sq = 0.f;
#pragma unroll
    for (int i = 0; i < 8; ++i) { s += v[i]; sq += v[i] * v[i]; }
#pragma unroll
    for (int o = 1; o < 32; o <<= 1) { s += __shfl_xor(s, o, 32); sq += __shfl_xor(sq, o, 32); }
    const float mean = s * (1.0f / CDIM);
    const float var  = sq * (1.0f / CDIM) - mean * mean;
    const float rstd = rsqrtf(var + 1e-5f);
    Pack8 pk;
#pragma unroll
    for (int i = 0; i < 8; ++i) {
        int c = lane * 8 + i;
        pk.h[i] = f2bf((v[i] - mean) * rstd * g[c] + bparm[c]);
    }
    size_t dst;
    if constexpr (DSTMODE == 0) {
        dst = (size_t)t * CDIM + lane * 8;
    } else {
        // cyclic shift by -4,-4 then window partition
        int bi = t >> 12, p = t & 4095;
        int hh = p >> 6, ww = p & 63;
        int hs = (hh + HIMG - SHIFTV) & 63, ws2 = (ww + WIMG - SHIFTV) & 63;
        int win = bi * NWIN + (hs >> 3) * 8 + (ws2 >> 3);
        int tok = (hs & 7) * 8 + (ws2 & 7);
        dst = ((size_t)win * NTOK + tok) * CDIM + lane * 8;
    }
    *(u32x4*)(out + dst) = pk.q;
}

// ---------------- generic 128x128-tile bf16 WMMA GEMM with fused epilogues ----------------
// A: MxK row-major bf16.  W: KxN row-major bf16.
// Fragments (CDNA5 16x16x32 bf16):
//   A: lane<16 row=lane, K={0..7,16..23}; lane>=16 same row set shifted by 8 -> two b128 LDS loads
//   B: lane = N col (mod 16), elems = K {0..7,16..23} (+8 upper lane half)  -> 16 u16 LDS loads
//   D: lane = N col (mod 16), VGPR r = row r (+8 for upper lane half)
// MODE 0: qkv scatter  1: proj + window-reverse + shortcut  2: fc1+GELU  3: fc2 += out
template <int MODE>
__global__ __launch_bounds__(256) void wmma_gemm_kernel(
    const u16* __restrict__ A, const u16* __restrict__ W,
    const float* __restrict__ bias, const int K,
    u16* __restrict__ outb, u16* __restrict__ auxk, u16* __restrict__ auxv,
    const float* __restrict__ shortcut, float* __restrict__ outf) {

    __shared__ __align__(16) u16 As[128][40];   // M x K, +8 pad (row stride 80B, 16B aligned)
    __shared__ __align__(16) u16 Bs[32][136];   // K x N, +8 pad (row stride 272B, 16B aligned)

    const int tid = threadIdx.x;
    const int wave = tid >> 5, lane = tid & 31;
    const int lm = lane & 15, hi = lane >> 4, kof = hi * 8;
    const int wM = (wave & 3) * 32, wN = (wave >> 2) * 64;
    const int Nn = gridDim.y * 128;

    v8f acc[2][4];
#pragma unroll
    for (int mt = 0; mt < 2; ++mt)
#pragma unroll
        for (int nt = 0; nt < 4; ++nt) acc[mt][nt] = vz8();

    const u16* pA = A + (size_t)blockIdx.x * 128 * K;
    const u16* pW = W + (size_t)blockIdx.y * 128;

    for (int kb = 0; kb < K; kb += 32) {
#pragma unroll
        for (int i = 0; i < 2; ++i) {
            int c = tid + i * 256;
            int ar = c >> 2, ao = (c & 3) * 8;                 // A tile: 128 rows x 4 chunks
            *(u32x4*)&As[ar][ao] = *(const u32x4*)(pA + (size_t)ar * K + kb + ao);
            int br = c >> 4, bo = (c & 15) * 8;                // B tile: 32 rows x 16 chunks
            *(u32x4*)&Bs[br][bo] = *(const u32x4*)(pW + (size_t)(kb + br) * Nn + bo);
        }
        __syncthreads();

        Frag a[2], b[4];
#pragma unroll
        for (int mt = 0; mt < 2; ++mt) {
            int r = wM + mt * 16 + lm;
            a[mt].q[0] = *(const u32x4*)&As[r][kof];
            a[mt].q[1] = *(const u32x4*)&As[r][16 + kof];
        }
#pragma unroll
        for (int nt = 0; nt < 4; ++nt) {
            int cb = wN + nt * 16;
#pragma unroll
            for (int s = 0; s < 16; ++s) b[nt].u[s] = Bs[(s >> 3) * 16 + kof + (s & 7)][cb + lm];
        }
#pragma unroll
        for (int mt = 0; mt < 2; ++mt)
#pragma unroll
            for (int nt = 0; nt < 4; ++nt)
                acc[mt][nt] = __builtin_amdgcn_wmma_f32_16x16x32_bf16(
                    false, a[mt].bf, false, b[nt].bf, (short)0, acc[mt][nt], false, false);
        __syncthreads();
    }

    // -------- epilogue --------
#pragma unroll
    for (int mt = 0; mt < 2; ++mt)
#pragma unroll
        for (int nt = 0; nt < 4; ++nt)
#pragma unroll
            for (int r = 0; r < 8; ++r) {
                int grow = blockIdx.x * 128 + wM + mt * 16 + r + hi * 8;
                int gcol = blockIdx.y * 128 + wN + nt * 16 + lm;
                float vv = acc[mt][nt][r] + bias[gcol];
                if constexpr (MODE == 0) {            // qkv -> q,k(T),v per-head layouts
                    int which = gcol >> 8, h = (gcol >> 5) & 7, d = gcol & 31;
                    int win = grow >> 6, tok = grow & 63;
                    size_t wh = (size_t)(win * NHEAD + h);
                    if (which == 0)      outb[(wh * NTOK + tok) * HDIM + d] = f2bf(vv * 0.176776695f);
                    else if (which == 1) auxk[(wh * HDIM + d) * NTOK + tok] = f2bf(vv);
                    else                 auxv[(wh * NTOK + tok) * HDIM + d] = f2bf(vv);
                } else if constexpr (MODE == 1) {     // proj: window reverse + unshift + shortcut
                    int win = grow >> 6, tok = grow & 63;
                    int bi = win >> 6, wi = win & 63;
                    int hs  = (wi >> 3) * 8 + (tok >> 3);
                    int wsx = (wi & 7) * 8 + (tok & 7);
                    int ho = (hs + SHIFTV) & 63, wo = (wsx + SHIFTV) & 63;
                    size_t idx = ((size_t)bi * (HIMG * WIMG) + ho * WIMG + wo) * CDIM + gcol;
                    outf[idx] = vv + shortcut[idx];
                } else if constexpr (MODE == 2) {     // fc1 + exact GELU
                    float ge = 0.5f * vv * (1.0f + erff(vv * 0.70710678f));
                    outb[(size_t)grow * Nn + gcol] = f2bf(ge);
                } else {                              // fc2 accumulate into residual
                    size_t idx = (size_t)grow * CDIM + gcol;
                    outf[idx] += vv;
                }
            }
}

// ---------------- windowed attention: one block per (window, head), 4 waves ----------------
__global__ __launch_bounds__(128) void attn_kernel(const u16* __restrict__ qb,
                                                   const u16* __restrict__ kb,
                                                   const u16* __restrict__ vb,
                                                   const float* __restrict__ btab,
                                                   u16* __restrict__ aout) {
    __shared__ __align__(16) u16 Kt[HDIM][72];      // K^T tile: d x tok
    __shared__ __align__(16) u16 Vs[NTOK][40];      // V tile:   tok x d
    __shared__ __align__(16) u16 Ps[4][16][72];     // softmax(P) per wave strip

    const int tid = threadIdx.x, wave = tid >> 5, lane = tid & 31;
    const int lm = lane & 15, hi = lane >> 4, kof = hi * 8;
    const int win = blockIdx.x, head = blockIdx.y;
    const size_t base = (size_t)(win * NHEAD + head) * (NTOK * HDIM);

#pragma unroll
    for (int i = 0; i < 2; ++i) {
        int c = tid + i * 128;
        int dd = c >> 3, doff = (c & 7) * 8;        // kb is [d][tok]
        *(u32x4*)&Kt[dd][doff] = *(const u32x4*)(kb + base + (size_t)dd * NTOK + doff);
        int n = c >> 2, noff = (c & 3) * 8;         // vb is [tok][d]
        *(u32x4*)&Vs[n][noff]  = *(const u32x4*)(vb + base + (size_t)n * HDIM + noff);
    }
    __syncthreads();

    // S = Q * K^T  (K dim = HDIM = 32 -> exactly one WMMA per 16x16 tile)
    Frag qf;
    const u16* qr = qb + base + (size_t)(wave * 16 + lm) * HDIM;
    qf.q[0] = *(const u32x4*)(qr + kof);
    qf.q[1] = *(const u32x4*)(qr + 16 + kof);

    v8f s[4];
    const v8f czero = vz8();
#pragma unroll
    for (int nt = 0; nt < 4; ++nt) {
        Frag kf;                                    // B: lane = tok col (mod 16), elems = d (K, table order)
#pragma unroll
        for (int s = 0; s < 16; ++s) kf.u[s] = Kt[(s >> 3) * 16 + kof + (s & 7)][nt * 16 + lm];
        s[nt] = __builtin_amdgcn_wmma_f32_16x16x32_bf16(
            false, qf.bf, false, kf.bf, (short)0, czero, false, false);
    }

    // relative-position bias + shifted-window mask (computed from indices)
    const int wi = win & 63, bwh = wi >> 3, bww = wi & 7;
    int mh[4], mw[4], rc[4];
#pragma unroll
    for (int nt = 0; nt < 4; ++nt) {
        int m = nt * 16 + lm;
        mh[nt] = m >> 3; mw[nt] = m & 7;
        rc[nt] = reg3(bwh * 8 + mh[nt]) * 3 + reg3(bww * 8 + mw[nt]);
    }
#pragma unroll
    for (int r = 0; r < 8; ++r) {
        int n = wave * 16 + r + hi * 8;
        int nh = n >> 3, nw = n & 7;
        int rr = reg3(bwh * 8 + nh) * 3 + reg3(bww * 8 + nw);
#pragma unroll
        for (int nt = 0; nt < 4; ++nt) {
            int bidx = ((nh - mh[nt] + 7) * 15 + (nw - mw[nt] + 7)) * NHEAD + head;
            s[nt][r] += btab[bidx] + ((rr == rc[nt]) ? 0.0f : -100.0f);
        }
    }

    // row softmax (row lives across one 16-lane half; cols split over 4 tiles)
#pragma unroll
    for (int r = 0; r < 8; ++r) {
        float mx = fmaxf(fmaxf(s[0][r], s[1][r]), fmaxf(s[2][r], s[3][r]));
#pragma unroll
        for (int o = 1; o < 16; o <<= 1) mx = fmaxf(mx, __shfl_xor(mx, o, 32));
        float e[4], sum = 0.f;
#pragma unroll
        for (int nt = 0; nt < 4; ++nt) { e[nt] = __expf(s[nt][r] - mx); sum += e[nt]; }
#pragma unroll
        for (int o = 1; o < 16; o <<= 1) sum += __shfl_xor(sum, o, 32);
        float inv = 1.0f / sum;
#pragma unroll
        for (int nt = 0; nt < 4; ++nt)
            Ps[wave][r + hi * 8][nt * 16 + lm] = f2bf(e[nt] * inv);
    }
    __syncthreads();

    // O = P * V  (K dim = 64 -> two WMMA steps)
    v8f o2[2];
    o2[0] = vz8(); o2[1] = vz8();
#pragma unroll
    for (int ks = 0; ks < NTOK; ks += 32) {
        Frag pf;                                    // A: lane = row, elems = K(tok)
        pf.q[0] = *(const u32x4*)&Ps[wave][lm][ks + kof];
        pf.q[1] = *(const u32x4*)&Ps[wave][lm][ks + 16 + kof];
#pragma unroll
        for (int nt = 0; nt < 2; ++nt) {
            Frag vf;                                // B: lane = d col (mod 16), elems = K(tok rel, table order)
#pragma unroll
            for (int s = 0; s < 16; ++s) vf.u[s] = Vs[ks + (s >> 3) * 16 + kof + (s & 7)][nt * 16 + lm];
            o2[nt] = __builtin_amdgcn_wmma_f32_16x16x32_bf16(
                false, pf.bf, false, vf.bf, (short)0, o2[nt], false, false);
        }
    }
#pragma unroll
    for (int nt = 0; nt < 2; ++nt)
#pragma unroll
        for (int r = 0; r < 8; ++r) {
            int row = win * NTOK + wave * 16 + r + hi * 8;
            int col = head * HDIM + nt * 16 + lm;
            aout[(size_t)row * CDIM + col] = f2bf(o2[nt][r]);
        }
}

// ---------------- host orchestration ----------------
extern "C" void kernel_launch(void* const* d_in, const int* in_sizes, int n_in,
                              void* d_out, int out_size, void* d_ws, size_t ws_size,
                              hipStream_t stream) {
    (void)in_sizes; (void)n_in; (void)out_size; (void)ws_size;
    const float* x      = (const float*)d_in[0];
    const float* n1g    = (const float*)d_in[1];
    const float* n1b    = (const float*)d_in[2];
    const float* qkv_w  = (const float*)d_in[3];
    const float* qkv_b  = (const float*)d_in[4];
    const float* proj_w = (const float*)d_in[5];
    const float* proj_b = (const float*)d_in[6];
    const float* btab   = (const float*)d_in[7];
    const float* n2g    = (const float*)d_in[8];
    const float* n2b    = (const float*)d_in[9];
    const float* fc1_w  = (const float*)d_in[10];
    const float* fc1_b  = (const float*)d_in[11];
    const float* fc2_w  = (const float*)d_in[12];
    const float* fc2_b  = (const float*)d_in[13];
    float* out = (float*)d_out;
    u16* ws = (u16*)d_ws;

    // weights -> bf16
    cvt_bf16_kernel<<<768,  256, 0, stream>>>(qkv_w,  ws + OFF_WQKV);
    cvt_bf16_kernel<<<256,  256, 0, stream>>>(proj_w, ws + OFF_WPROJ);
    cvt_bf16_kernel<<<1024, 256, 0, stream>>>(fc1_w,  ws + OFF_WFC1);
    cvt_bf16_kernel<<<1024, 256, 0, stream>>>(fc2_w,  ws + OFF_WFC2);

    // LN1 + cyclic shift + window partition
    ln_kernel<1><<<MROWS / 8, 256, 0, stream>>>(x, n1g, n1b, ws + OFF_XW);

    // QKV GEMM (131072 x 768, K=256), scatter q(scaled)/k(T)/v
    wmma_gemm_kernel<0><<<dim3(MROWS / 128, 6), 256, 0, stream>>>(
        ws + OFF_XW, ws + OFF_WQKV, qkv_b, CDIM,
        ws + OFF_Q, ws + OFF_K, ws + OFF_V, nullptr, nullptr);

    // windowed multi-head attention -> attn_out (reuses XW region)
    attn_kernel<<<dim3(BW, NHEAD), 128, 0, stream>>>(
        ws + OFF_Q, ws + OFF_K, ws + OFF_V, btab, ws + OFF_XW);

    // proj GEMM + window reverse + unshift + residual -> d_out (fp32 x_res)
    wmma_gemm_kernel<1><<<dim3(MROWS / 128, 2), 256, 0, stream>>>(
        ws + OFF_XW, ws + OFF_WPROJ, proj_b, CDIM,
        nullptr, nullptr, nullptr, x, out);

    // LN2 on residual
    ln_kernel<0><<<MROWS / 8, 256, 0, stream>>>(out, n2g, n2b, ws + OFF_Q);

    // fc1 + GELU (131072 x 1024, K=256)
    wmma_gemm_kernel<2><<<dim3(MROWS / 128, 8), 256, 0, stream>>>(
        ws + OFF_Q, ws + OFF_WFC1, fc1_b, CDIM,
        ws + OFF_G1, nullptr, nullptr, nullptr, nullptr);

    // fc2 (131072 x 256, K=1024) accumulated into residual
    wmma_gemm_kernel<3><<<dim3(MROWS / 128, 2), 256, 0, stream>>>(
        ws + OFF_G1, ws + OFF_WFC2, fc2_b, 4 * CDIM,
        nullptr, nullptr, nullptr, nullptr, out);
}
